// CrossModalAttention_10711648436323
// MI455X (gfx1250) — matrix operand order repaired and checked by reading, not compiled, not run on hardware
//
#include <hip/hip_runtime.h>
#include <hip/hip_bf16.h>

typedef __bf16 bf16_t;
typedef __bf16 v16bf __attribute__((ext_vector_type(16)));
typedef float  v8f   __attribute__((ext_vector_type(8)));
typedef float  v4f   __attribute__((ext_vector_type(4)));
typedef unsigned int v4u __attribute__((ext_vector_type(4)));
typedef unsigned int v2u __attribute__((ext_vector_type(2)));

// ---------------------------------------------------------------------------
// Fragment helpers (wave32; C/D per CDNA5 ISA §7.12.2, A/B use contiguous k)
//   A 16x32 bf16 : lane L -> row M = L&15 ; elems a[j] = K (16*hf + j)
//   B 32x16 bf16 : lane L -> col  = L&15 ; elems b[j] = K (16*hf + j)
//   C/D 16x16 f32: lane L -> col  = L&15 ; VGPR r -> row M = r + 8*hf
// ---------------------------------------------------------------------------
__device__ __forceinline__ v16bf load_frag2(const bf16_t* p0, const bf16_t* p1) {
  union U { v16bf v; v4u q[2]; } u;
  u.q[0] = *reinterpret_cast<const v4u*>(p0);   // 16B aligned chunk -> elems 0..7
  u.q[1] = *reinterpret_cast<const v4u*>(p1);   // 16B aligned chunk -> elems 8..15
  return u.v;
}

__device__ __forceinline__ v8f wmma_bf16(v16bf a, v16bf b, v8f c) {
  return __builtin_amdgcn_wmma_f32_16x16x32_bf16(false, a, false, b, (short)0, c,
                                                 false, false);
}

#define NN 4096
#define CC 256

// ---------------------------------------------------------------------------
// Kernel 1: fused Q/K/V 1x1-conv projections as bf16 WMMA GEMM.
//   Out[o,n] = sum_c W[o,c] * X[c,n] + bias[o]
// Block = 256 thr (8 waves), tile 64o x 64n, K-loop over C in steps of 32.
// Q,K stored [b,h,N,d] (d contiguous); V stored [b,C,N] (n contiguous).
// Staging uses uniform fully-unrolled trip counts + float4 global loads so
// loads pipeline (no per-element s_wait_loadcnt 0).
// ---------------------------------------------------------------------------
__global__ __launch_bounds__(256) void qkv_proj_kernel(
    const float* __restrict__ qf, const float* __restrict__ kf,
    const float* __restrict__ q_w, const float* __restrict__ q_b,
    const float* __restrict__ k_w, const float* __restrict__ k_b,
    const float* __restrict__ v_w, const float* __restrict__ v_b,
    bf16_t* __restrict__ Qbuf, bf16_t* __restrict__ Kbuf,
    bf16_t* __restrict__ Vbuf) {
  __shared__ bf16_t XT[64][40];   // X tile transposed [n][c], row stride 80B (16B aligned)
  __shared__ bf16_t WT[64][32];   // W tile [o][c], row stride 64B

  const int n0   = blockIdx.x * 64;
  const int proj = blockIdx.y >> 2;          // 0=Q 1=K 2=V
  const int o0   = (blockIdx.y & 3) * 64;
  const int b    = blockIdx.z;
  const float* X  = (proj == 0 ? qf : kf) + (size_t)b * CC * NN;
  const float* W  = proj == 0 ? q_w : (proj == 1 ? k_w : v_w);
  const float* Bs = proj == 0 ? q_b : (proj == 1 ? k_b : v_b);

  const int tid = threadIdx.x, lane = tid & 31, w = tid >> 5;
  const int hf = lane >> 4, l16 = lane & 15;
  const int o_sub = w & 3;                   // A-row subtile (16 o rows)
  const int n_sub = (w >> 2) * 2;            // first of two B-col subtiles

  // staging coordinates (all uniform trip counts)
  const int xc4 = (tid & 15) * 4;            // n-offset of this thread's float4
  const int xr  = tid >> 4;                  // c row (0..15), +16 on 2nd pass
  const int wj4 = (tid & 7) * 4;             // c-offset of this thread's float4
  const int wi  = tid >> 3;                  // o row (0..31), +32 on 2nd pass

  v8f acc[2] = {};
  for (int kk = 0; kk < CC; kk += 32) {
    // --- stage X^T: 2 passes x float4 load, 4 scattered bf16 LDS stores ---
#pragma unroll
    for (int p = 0; p < 2; ++p) {
      const int row = xr + p * 16;
      const v4f xv = *reinterpret_cast<const v4f*>(&X[(size_t)(kk + row) * NN + n0 + xc4]);
#pragma unroll
      for (int e = 0; e < 4; ++e) XT[xc4 + e][row] = (bf16_t)xv[e];
    }
    // --- stage W: 2 passes x float4 load, packed b64 LDS store ---
#pragma unroll
    for (int p = 0; p < 2; ++p) {
      const int row = wi + p * 32;
      const v4f wv = *reinterpret_cast<const v4f*>(&W[(size_t)(o0 + row) * CC + kk + wj4]);
      union { bf16_t e[4]; v2u u; } pk;
#pragma unroll
      for (int e = 0; e < 4; ++e) pk.e[e] = (bf16_t)wv[e];
      *reinterpret_cast<v2u*>(&WT[row][wj4]) = pk.u;
    }
    __syncthreads();
    const int o_l = o_sub * 16 + l16;
    const v16bf a = load_frag2(&WT[o_l][16 * hf], &WT[o_l][16 * hf + 8]);
#pragma unroll
    for (int t = 0; t < 2; ++t) {
      const int n_l = (n_sub + t) * 16 + l16;
      const int kb  = 16 * hf;
      const v16bf bb = load_frag2(&XT[n_l][kb], &XT[n_l][kb + 8]);
      acc[t] = wmma_bf16(a, bb, acc[t]);
    }
    __syncthreads();
  }
  // epilogue: bias + layout-specific bf16 store
#pragma unroll
  for (int r = 0; r < 8; ++r) {
    const int o_g = o0 + o_sub * 16 + r + 8 * hf;
    const float bias = Bs[o_g];
#pragma unroll
    for (int t = 0; t < 2; ++t) {
      const int n = n0 + (n_sub + t) * 16 + l16;
      const float v = acc[t][r] + bias;
      if (proj == 2) {
        Vbuf[((size_t)b * CC + o_g) * NN + n] = (bf16_t)v;
      } else {
        const int head = o_g >> 5, dd = o_g & 31;
        bf16_t* dst = (proj == 0) ? Qbuf : Kbuf;
        dst[(((size_t)(b * 8 + head)) * NN + n) * 32 + dd] = (bf16_t)v;
      }
    }
  }
}

// ---------------------------------------------------------------------------
// Kernel 2: fused flash attention. One block = 128 query rows of one (b,h);
// each wave owns 16 rows (one A tile), Q fragment resident in VGPRs.
// Per 32-key chunk: 2 WMMA (S) -> online softmax (shfl butterflies within
// 16-lane halves) -> P through per-wave LDS (C->A layout) -> 2 WMMA (P·V).
// Next chunk's K/V rows are prefetched (global_prefetch_b8) to hide latency.
// ---------------------------------------------------------------------------
__global__ __launch_bounds__(256) void attention_kernel(
    const bf16_t* __restrict__ Qbuf, const bf16_t* __restrict__ Kbuf,
    const bf16_t* __restrict__ Vbuf, bf16_t* __restrict__ Obuf) {
  __shared__ bf16_t Pt[8][16][32];           // per-wave P staging (1KB each)

  const int tid = threadIdx.x, lane = tid & 31, w = tid >> 5;
  const int hf = lane >> 4, l16 = lane & 15;
  const int n0 = blockIdx.x * 128;
  const int h = blockIdx.y, b = blockIdx.z, bh = b * 8 + h;
  const bf16_t* Qb = Qbuf + (size_t)bh * NN * 32;
  const bf16_t* Kb = Kbuf + (size_t)bh * NN * 32;
  const bf16_t* Vb = Vbuf + ((size_t)b * CC + h * 32) * NN;

  const int nrow = n0 + w * 16 + l16;        // this lane's A-matrix row
  const v16bf qa = load_frag2(&Qb[(size_t)nrow * 32 + 16 * hf],
                              &Qb[(size_t)nrow * 32 + 16 * hf + 8]);

  v8f acc0 = {}, acc1 = {};                  // O accum, dv 0..15 / 16..31
  float mrow[8], lrow[8];                    // stats for rows r+8*hf (replicated/half)
#pragma unroll
  for (int r = 0; r < 8; ++r) { mrow[r] = -3.0e38f; lrow[r] = 0.f; }
  const float scale = 0.17677669529663687f;  // 1/sqrt(32)

  for (int mb = 0; mb < NN; mb += 32) {
    // prefetch next chunk (K rows + V row segments) while computing this one
    if (mb + 32 < NN) {
      __builtin_prefetch(&Kb[(size_t)(mb + 32 + lane) * 32], 0, 3);
      __builtin_prefetch(&Vb[(size_t)lane * NN + mb + 32], 0, 3);
    }
    v8f s[2];
#pragma unroll
    for (int t = 0; t < 2; ++t) {            // S = Q * K^T (two 16-col tiles)
      const int m = mb + t * 16 + l16;
      const v16bf kfr = load_frag2(&Kb[(size_t)m * 32 + 16 * hf],
                                   &Kb[(size_t)m * 32 + 16 * hf + 8]);
      v8f z = {};
      s[t] = wmma_bf16(qa, kfr, z);
    }
    float p0[8], p1[8], sf[8];
#pragma unroll
    for (int r = 0; r < 8; ++r) {            // online softmax per row
      const float a0 = s[0][r] * scale, a1 = s[1][r] * scale;
      float mx = fmaxf(a0, a1);
#pragma unroll
      for (int msk = 1; msk < 16; msk <<= 1) mx = fmaxf(mx, __shfl_xor(mx, msk, 32));
      const float mnew = fmaxf(mrow[r], mx);
      const float f = __expf(mrow[r] - mnew);
      const float e0 = __expf(a0 - mnew), e1 = __expf(a1 - mnew);
      float rs = e0 + e1;
#pragma unroll
      for (int msk = 1; msk < 16; msk <<= 1) rs += __shfl_xor(rs, msk, 32);
      lrow[r] = lrow[r] * f + rs;
      mrow[r] = mnew; sf[r] = f; p0[r] = e0; p1[r] = e1;
    }
#pragma unroll
    for (int r = 0; r < 8; ++r) { acc0[r] *= sf[r]; acc1[r] *= sf[r]; }
#pragma unroll
    for (int r = 0; r < 8; ++r) {            // C-layout -> LDS (wave-private)
      const int M = r + 8 * hf;
      Pt[w][M][l16]      = (bf16_t)p0[r];
      Pt[w][M][16 + l16] = (bf16_t)p1[r];
    }
    __asm__ volatile("s_wait_dscnt 0x0" ::: "memory");
    const v16bf pa = load_frag2(&Pt[w][l16][16 * hf], &Pt[w][l16][16 * hf + 8]);
#pragma unroll
    for (int t = 0; t < 2; ++t) {            // O += P * V
      const int dv = t * 16 + l16;
      const bf16_t* vp = &Vb[(size_t)dv * NN + mb + 16 * hf];
      const v16bf vf = load_frag2(vp, vp + 8);
      if (t == 0) acc0 = wmma_bf16(pa, vf, acc0);
      else        acc1 = wmma_bf16(pa, vf, acc1);
    }
    __asm__ volatile("" ::: "memory");       // keep next iter's LDS writes behind reads
  }
  // epilogue: O = acc/l ; store bf16 [b, h*32+dv, n] (8 contiguous n per lane)
#pragma unroll
  for (int t = 0; t < 2; ++t) {
    const int dv = t * 16 + l16;
    union { v4u q; bf16_t e[8]; } pk;
    const v8f& ac = (t == 0) ? acc0 : acc1;
#pragma unroll
    for (int j = 0; j < 8; ++j) pk.e[j] = (bf16_t)(ac[j] / lrow[j]);
    bf16_t* dst = &Obuf[((size_t)b * CC + h * 32 + dv) * NN + n0 + w * 16 + 8 * hf];
    *reinterpret_cast<v4u*>(dst) = pk.q;
  }
}

// ---------------------------------------------------------------------------
// Kernel 3: output projection WMMA GEMM + residual add + GroupNorm partial
// sums (LDS ds_add -> one global atomic per group per block).
// ---------------------------------------------------------------------------
__global__ void zero_stats_kernel(float* __restrict__ stats) {
  stats[threadIdx.x] = 0.f;
}

__global__ __launch_bounds__(256) void outproj_kernel(
    const bf16_t* __restrict__ Obuf, const float* __restrict__ out_w,
    const float* __restrict__ out_b, const float* __restrict__ qf,
    float* __restrict__ Y, float* __restrict__ stats) {
  __shared__ bf16_t XT[64][40];
  __shared__ bf16_t WT[64][32];
  __shared__ float gsum[8], gsq[8];

  const int tid = threadIdx.x, lane = tid & 31, w = tid >> 5;
  const int hf = lane >> 4, l16 = lane & 15;
  const int n0 = blockIdx.x * 64, c0 = blockIdx.y * 64, b = blockIdx.z;
  const bf16_t* Xb = Obuf + (size_t)b * CC * NN;
  if (tid < 8) { gsum[tid] = 0.f; gsq[tid] = 0.f; }

  const int o_sub = w & 3, n_sub = (w >> 2) * 2;
  const int xn8 = (tid & 7) * 8;             // n-offset of this thread's 8-bf16 chunk
  const int xr  = tid >> 3;                  // o row (0..31), single pass
  const int wj4 = (tid & 7) * 4;
  const int wi  = tid >> 3;

  v8f acc[2] = {};
  for (int kk = 0; kk < CC; kk += 32) {
    // --- stage O^T (bf16 already): one b128 load, 8 scattered bf16 stores ---
    {
      union { v4u q; bf16_t e[8]; } pk;
      pk.q = *reinterpret_cast<const v4u*>(&Xb[(size_t)(kk + xr) * NN + n0 + xn8]);
#pragma unroll
      for (int e = 0; e < 8; ++e) XT[xn8 + e][xr] = pk.e[e];
    }
    // --- stage out_w: 2 passes x float4 load, packed b64 store ---
#pragma unroll
    for (int p = 0; p < 2; ++p) {
      const int row = wi + p * 32;
      const v4f wv = *reinterpret_cast<const v4f*>(&out_w[(size_t)(c0 + row) * CC + kk + wj4]);
      union { bf16_t e[4]; v2u u; } pk;
#pragma unroll
      for (int e = 0; e < 4; ++e) pk.e[e] = (bf16_t)wv[e];
      *reinterpret_cast<v2u*>(&WT[row][wj4]) = pk.u;
    }
    __syncthreads();
    const int c_l = o_sub * 16 + l16;
    const v16bf a = load_frag2(&WT[c_l][16 * hf], &WT[c_l][16 * hf + 8]);
#pragma unroll
    for (int t = 0; t < 2; ++t) {
      const int n_l = (n_sub + t) * 16 + l16;
      const int kb  = 16 * hf;
      const v16bf bb = load_frag2(&XT[n_l][kb], &XT[n_l][kb + 8]);
      acc[t] = wmma_bf16(a, bb, acc[t]);
    }
    __syncthreads();
  }
#pragma unroll
  for (int r = 0; r < 8; ++r) {
    const int c = c0 + o_sub * 16 + r + 8 * hf;
    const float bias = out_b[c];
#pragma unroll
    for (int t = 0; t < 2; ++t) {
      const int n = n0 + (n_sub + t) * 16 + l16;
      const size_t idx = ((size_t)b * CC + c) * NN + n;
      const float y = acc[t][r] + bias + qf[idx];
      Y[idx] = y;
      atomicAdd(&gsum[(c - c0) >> 3], y);
      atomicAdd(&gsq[(c - c0) >> 3], y * y);
    }
  }
  __syncthreads();
  if (tid < 8) {
    const int g = (c0 >> 3) + tid;
    atomicAdd(&stats[(b * 32 + g) * 2 + 0], gsum[tid]);
    atomicAdd(&stats[(b * 32 + g) * 2 + 1], gsq[tid]);
  }
}

// ---------------------------------------------------------------------------
// Kernel 4: GroupNorm finalize, in-place on d_out (float4). One block/(b,g).
// ---------------------------------------------------------------------------
__global__ __launch_bounds__(256) void gnorm_kernel(
    float* __restrict__ Y, const float* __restrict__ stats,
    const float* __restrict__ gn_w, const float* __restrict__ gn_b) {
  const int bg = blockIdx.x;               // 0..63
  const int b = bg >> 5, g = bg & 31;
  const float inv = 1.0f / (8.0f * NN);
  const float mean = stats[bg * 2 + 0] * inv;
  const float var  = stats[bg * 2 + 1] * inv - mean * mean;
  const float rinv = rsqrtf(var + 1e-5f);
  const size_t base = ((size_t)b * CC + g * 8) * NN;
  for (int i4 = threadIdx.x; i4 < (8 * NN) / 4; i4 += 256) {
    const int i = i4 * 4;
    const int c = g * 8 + (i >> 12);       // 4 | 4096, so same c for all 4 lanes of v4
    const float sw = rinv * gn_w[c];
    const float sb = gn_b[c] - mean * sw;
    v4f y = *reinterpret_cast<v4f*>(&Y[base + i]);
#pragma unroll
    for (int e = 0; e < 4; ++e) y[e] = y[e] * sw + sb;
    *reinterpret_cast<v4f*>(&Y[base + i]) = y;
  }
}

// ---------------------------------------------------------------------------
extern "C" void kernel_launch(void* const* d_in, const int* in_sizes, int n_in,
                              void* d_out, int out_size, void* d_ws, size_t ws_size,
                              hipStream_t stream) {
  const float* qf    = (const float*)d_in[0];
  const float* kf    = (const float*)d_in[1];
  const float* q_w   = (const float*)d_in[2];
  const float* q_b   = (const float*)d_in[3];
  const float* k_w   = (const float*)d_in[4];
  const float* k_b   = (const float*)d_in[5];
  const float* v_w   = (const float*)d_in[6];
  const float* v_b   = (const float*)d_in[7];
  const float* out_w = (const float*)d_in[8];
  const float* out_b = (const float*)d_in[9];
  const float* gn_w  = (const float*)d_in[10];
  const float* gn_b  = (const float*)d_in[11];

  char* ws = (char*)d_ws;
  bf16_t* Qbuf = (bf16_t*)(ws + (size_t)0);          // 4 MiB  [b,h,N,d]
  bf16_t* Kbuf = (bf16_t*)(ws + ((size_t)4  << 20)); // 4 MiB  [b,h,N,d]
  bf16_t* Vbuf = (bf16_t*)(ws + ((size_t)8  << 20)); // 4 MiB  [b,C,N]
  bf16_t* Obuf = (bf16_t*)(ws + ((size_t)12 << 20)); // 4 MiB  [b,C,N]
  float*  stats = (float*)(ws + ((size_t)16 << 20)); // 128 floats
  float*  Y = (float*)d_out;

  qkv_proj_kernel<<<dim3(64, 12, 2), 256, 0, stream>>>(
      qf, kf, q_w, q_b, k_w, k_b, v_w, v_b, Qbuf, Kbuf, Vbuf);
  attention_kernel<<<dim3(32, 8, 2), 256, 0, stream>>>(Qbuf, Kbuf, Vbuf, Obuf);
  zero_stats_kernel<<<1, 128, 0, stream>>>(stats);
  outproj_kernel<<<dim3(64, 4, 2), 256, 0, stream>>>(Obuf, out_w, out_b, qf, Y, stats);
  gnorm_kernel<<<64, 256, 0, stream>>>(Y, stats, gn_w, gn_b);
}
